// ExpertCommunicationLayer_4252017623350
// MI455X (gfx1250) — matrix operand order repaired and checked by reading, not compiled, not run on hardware
//
#include <hip/hip_runtime.h>
#include <stdint.h>

// Problem dims (fixed by the reference's setup_inputs).
#define B_SZ   2048
#define E_SZ   16
#define H_SZ   1024
#define NH_SZ  8
#define HD_SZ  128

typedef __attribute__((ext_vector_type(16))) __bf16 v16bf;
typedef __attribute__((ext_vector_type(8)))  float  v8f;
typedef __attribute__((ext_vector_type(4)))  int    v4i;

union BF16Frag { v16bf v; uint4 q[2]; unsigned short u[16]; };

#if __has_builtin(__builtin_amdgcn_global_load_async_to_lds_b128) && \
    __has_builtin(__builtin_amdgcn_s_wait_asynccnt)
#define GEMM_ASYNC 1
#else
#define GEMM_ASYNC 0
#endif
#define AS1G __attribute__((address_space(1)))
#define AS3L __attribute__((address_space(3)))

__device__ __forceinline__ unsigned short f2bf(float f) {
  union { float f; unsigned int u; } c; c.f = f;
  unsigned int r = c.u + 0x7FFFu + ((c.u >> 16) & 1u);   // RNE
  return (unsigned short)(r >> 16);
}
__device__ __forceinline__ float bf2f(unsigned short b) {
  union { unsigned int u; float f; } c; c.u = ((unsigned int)b) << 16;
  return c.f;
}

// ---------------------------------------------------------------------------
// 1) fp32 -> bf16 weight conversion (wq,wk,wv,wo selected by blockIdx.z)
// ---------------------------------------------------------------------------
__global__ __launch_bounds__(256) void wconv_kernel(
    const float* __restrict__ wq, const float* __restrict__ wk,
    const float* __restrict__ wv, const float* __restrict__ wo,
    unsigned short* __restrict__ dst) {
  const float* src = (blockIdx.z == 0) ? wq : (blockIdx.z == 1) ? wk
                     : (blockIdx.z == 2) ? wv : wo;
  size_t base = ((size_t)blockIdx.x * blockDim.x + threadIdx.x) * 4;
  float4 v = *(const float4*)(src + base);
  uint2 p;
  p.x = (unsigned)f2bf(v.x) | ((unsigned)f2bf(v.y) << 16);
  p.y = (unsigned)f2bf(v.z) | ((unsigned)f2bf(v.w) << 16);
  *(uint2*)(dst + (size_t)blockIdx.z * (size_t)H_SZ * H_SZ + base) = p;
}

// ---------------------------------------------------------------------------
// 2) comm = 0.5*(x[e-1]+x[e+1]) then LayerNorm.  One wave32 per (b,e) row.
// ---------------------------------------------------------------------------
__global__ __launch_bounds__(256) void commln_kernel(
    const float* __restrict__ x, const float* __restrict__ g,
    const float* __restrict__ beta, float* __restrict__ comm,
    unsigned short* __restrict__ xn) {
  int w = threadIdx.x >> 5, lane = threadIdx.x & 31;
  int row = blockIdx.x * 8 + w;
  int e = row & (E_SZ - 1);
  const float* xr = x + (size_t)row * H_SZ;
  float cv[32];
  float s = 0.f, s2 = 0.f;
#pragma unroll
  for (int i = 0; i < 32; ++i) {
    int hh = i * 32 + lane;
    float c = 0.f;
    if (e > 0)        c += xr[hh - H_SZ];
    if (e < E_SZ - 1) c += xr[hh + H_SZ];
    c *= 0.5f;
    cv[i] = c; s += c; s2 += c * c;
    comm[(size_t)row * H_SZ + hh] = c;
  }
#pragma unroll
  for (int m = 16; m >= 1; m >>= 1) {
    s  += __shfl_xor(s,  m, 32);
    s2 += __shfl_xor(s2, m, 32);
  }
  float mean = s * (1.f / H_SZ);
  float var  = s2 * (1.f / H_SZ) - mean * mean;
  float rs   = rsqrtf(var + 1e-5f);
#pragma unroll
  for (int i = 0; i < 32; ++i) {
    int hh = i * 32 + lane;
    float v = (cv[i] - mean) * rs * g[hh] + beta[hh];
    xn[(size_t)row * H_SZ + hh] = f2bf(v);
  }
}

// ---------------------------------------------------------------------------
// bf16 WMMA GEMM core:  C[m,n] = sum_k A[m,k]*W[n,k] + bias[n] (+res[m,n])
// Block: 256 threads (8 waves). Block tile 256x128, wave tile 64x64
// (4x4 WMMA tiles, 16 ds_load_b128 : 16 v_wmma per K-step of 32).
// Double-buffered LDS; global->LDS via async-to-LDS when available.
// K loop is kept rolled (#pragma unroll 1) so only ONE iteration's fragments
// are live: acc(128) + B-frags(32) + A-frag(8) stays in the low-256 VGPR
// window (no vgpr-msb juggling / accumulator copy storm).
// ---------------------------------------------------------------------------
template <bool OUT_BF16, bool ADD_RES>
__device__ __forceinline__ void gemm_tile(
    const unsigned short* __restrict__ A,   // [M,1024] bf16
    const unsigned short* __restrict__ W,   // [1024,1024] bf16 (row n, contiguous k)
    const float* __restrict__ bias,         // [1024]
    const float* res,                       // [M,1024] (may alias outF)
    unsigned short* __restrict__ outB,      // bf16 out
    float* outF) {                          // f32 out
  __shared__ uint4 As[2][256 * 4];          // 2 x 16 KB  (256 rows x 32 bf16)
  __shared__ uint4 Bs[2][128 * 4];          // 2 x  8 KB  (128 rows x 32 bf16)
  const int K = H_SZ, N = H_SZ, KSTEPS = H_SZ / 32;
  const int t = threadIdx.x, lane = t & 31, w = t >> 5;
  const int m0 = blockIdx.y * 256, n0 = blockIdx.x * 128;
  const int wm = (w & 3) * 64, wn = (w >> 2) * 64;
  const int hi = lane >> 4, lo = lane & 15;

  // Fill assignment: thread t owns A row (m0+t) (64B/step) and half of W row
  // (n0 + t/2) (32B/step).
  const unsigned short* gA = A + (size_t)(m0 + t) * K;
  const unsigned short* gB = W + (size_t)(n0 + (t >> 1)) * K + (t & 1) * 16;
  const int aq0 = t * 4;                       // quad base in As
  const int bq0 = (t >> 1) * 4 + (t & 1) * 2;  // quad base in Bs

  v8f acc[4][4] = {};

#if GEMM_ASYNC
  auto fill = [&](int buf, int k0) {
#pragma unroll
    for (int qd = 0; qd < 4; ++qd)
      __builtin_amdgcn_global_load_async_to_lds_b128(
          (AS1G v4i*)(gA + k0 + qd * 8), (AS3L v4i*)&As[buf][aq0 + qd], 0, 0);
#pragma unroll
    for (int qd = 0; qd < 2; ++qd)
      __builtin_amdgcn_global_load_async_to_lds_b128(
          (AS1G v4i*)(gB + k0 + qd * 8), (AS3L v4i*)&Bs[buf][bq0 + qd], 0, 0);
  };
  fill(0, 0);
  __builtin_amdgcn_s_wait_asynccnt(0);
  __syncthreads();
#else
  uint4 ra[4], rb[2];
  auto gload = [&](int k0) {
#pragma unroll
    for (int qd = 0; qd < 4; ++qd) ra[qd] = *(const uint4*)(gA + k0 + qd * 8);
#pragma unroll
    for (int qd = 0; qd < 2; ++qd) rb[qd] = *(const uint4*)(gB + k0 + qd * 8);
    __builtin_prefetch(gA + k0 + 32, 0, 1);
    __builtin_prefetch(gB + k0 + 32, 0, 1);
  };
  auto lstore = [&](int buf) {
#pragma unroll
    for (int qd = 0; qd < 4; ++qd) As[buf][aq0 + qd] = ra[qd];
#pragma unroll
    for (int qd = 0; qd < 2; ++qd) Bs[buf][bq0 + qd] = rb[qd];
  };
  gload(0);
  lstore(0);
  __syncthreads();
#endif

#pragma unroll 1
  for (int kb = 0; kb < KSTEPS; ++kb) {
    const int cur = kb & 1;
    // Start refill of the other buffer; it hides under the WMMA burst.
#if GEMM_ASYNC
    if (kb + 1 < KSTEPS) fill(cur ^ 1, (kb + 1) * 32);
#else
    if (kb + 1 < KSTEPS) gload((kb + 1) * 32);
#endif

    const uint4* Asb = As[cur];
    const uint4* Bsb = Bs[cur];

    // B-frags stay resident (4 x 8 VGPRs); A-frags streamed per mt.
    BF16Frag bf[4];
#pragma unroll
    for (int nt = 0; nt < 4; ++nt) {   // B 32x16: lane col = n, lanes0-15 K0-7/16-23, lanes16-31 K8-15/24-31
      int rl = wn + nt * 16 + lo;
      bf[nt].q[0] = Bsb[rl * 4 + hi];
      bf[nt].q[1] = Bsb[rl * 4 + hi + 2];
    }
#pragma unroll
    for (int mt = 0; mt < 4; ++mt) {   // A 16x32: lanes0-15 K0-7/16-23, lanes16-31 K8-15/24-31
      BF16Frag af;
      int rl = wm + mt * 16 + lo;
      af.q[0] = Asb[rl * 4 + hi];
      af.q[1] = Asb[rl * 4 + hi + 2];
#pragma unroll
      for (int nt = 0; nt < 4; ++nt)
        acc[mt][nt] = __builtin_amdgcn_wmma_f32_16x16x32_bf16(
            false, af.v, false, bf[nt].v, (short)0, acc[mt][nt], false, false);
    }

    if (kb + 1 < KSTEPS) {
#if GEMM_ASYNC
      __builtin_amdgcn_s_wait_asynccnt(0);
#else
      lstore(cur ^ 1);
#endif
      __syncthreads();
    }
  }

#pragma unroll
  for (int nt = 0; nt < 4; ++nt) {
    int n = n0 + wn + nt * 16 + lo;
    float bv = bias[n];
#pragma unroll
    for (int mt = 0; mt < 4; ++mt) {
      int mbase = m0 + wm + mt * 16 + hi * 8;  // D: lanes0-15 M=r, lanes16-31 M=r+8
#pragma unroll
      for (int r = 0; r < 8; ++r) {
        size_t idx = (size_t)(mbase + r) * N + n;
        float v = acc[mt][nt][r] + bv;
        if (ADD_RES) v += res[idx];
        if (OUT_BF16) outB[idx] = f2bf(v);
        else          outF[idx] = v;
      }
    }
  }
}

__global__ __launch_bounds__(256) void gemm_qkv_kernel(
    const unsigned short* __restrict__ xn, const unsigned short* __restrict__ wbase,
    const float* __restrict__ bq, const float* __restrict__ bk, const float* __restrict__ bv,
    unsigned short* __restrict__ q, unsigned short* __restrict__ k,
    unsigned short* __restrict__ v) {
  int z = blockIdx.z;
  const unsigned short* W = wbase + (size_t)z * H_SZ * H_SZ;
  const float* bias = (z == 0) ? bq : (z == 1) ? bk : bv;
  unsigned short* out = (z == 0) ? q : (z == 1) ? k : v;
  gemm_tile<true, false>(xn, W, bias, nullptr, out, nullptr);
}

__global__ __launch_bounds__(256) void gemm_out_kernel(
    const unsigned short* __restrict__ att, const unsigned short* __restrict__ wo,
    const float* __restrict__ bo, const float* resid, float* out) {
  gemm_tile<false, true>(att, wo, bo, resid, nullptr, out);  // resid may alias out
}

// ---------------------------------------------------------------------------
// 4) Attention: one wave per (b,head).  scores via bf16 WMMA (K=128),
//    half-wave softmax, attn@v via f32 WMMA 16x16x4 (bf16 fallback).
// ---------------------------------------------------------------------------
__global__ __launch_bounds__(128) void attn_kernel(
    const unsigned short* __restrict__ q, const unsigned short* __restrict__ k,
    const unsigned short* __restrict__ v, unsigned short* att) {
  __shared__ float atile[4][16][16];
  const int lane = threadIdx.x & 31, w = threadIdx.x >> 5;
  const int pair = blockIdx.x * 4 + w;
  const int b = pair >> 3, h = pair & 7;
  const size_t base = (size_t)b * E_SZ * H_SZ + (size_t)h * HD_SZ;
  const unsigned short* qb = q + base;
  const unsigned short* kb = k + base;
  const unsigned short* vb = v + base;
  const int m = lane & 15, hi = lane >> 4;

  // scores[16x16] = q[16x128] @ k^T, K in 4 chunks of 32
  v8f sc = {};
#pragma unroll
  for (int kc = 0; kc < 4; ++kc) {
    BF16Frag aq, bk2;
    const unsigned short* ap = qb + (size_t)m * H_SZ + kc * 32 + hi * 8;
    aq.q[0] = *(const uint4*)ap;
    aq.q[1] = *(const uint4*)(ap + 16);
    const unsigned short* bp = kb + (size_t)m * H_SZ + kc * 32 + hi * 8;
    bk2.q[0] = *(const uint4*)bp;
    bk2.q[1] = *(const uint4*)(bp + 16);
    sc = __builtin_amdgcn_wmma_f32_16x16x32_bf16(
        false, aq.v, false, bk2.v, (short)0, sc, false, false);
  }

  // softmax over keys: row r+hi*8 lives in a 16-lane half -> shfl_xor 1..8
  const float scale = 0.0883883476483184f;  // 1/sqrt(128)
#pragma unroll
  for (int r = 0; r < 8; ++r) {
    float x0 = sc[r] * scale;
    float mx = x0;
#pragma unroll
    for (int s = 8; s >= 1; s >>= 1) mx = fmaxf(mx, __shfl_xor(mx, s, 32));
    float p = __expf(x0 - mx);
    float sm = p;
#pragma unroll
    for (int s = 8; s >= 1; s >>= 1) sm += __shfl_xor(sm, s, 32);
    atile[w][r + hi * 8][m] = p / sm;   // same-wave LDS: in-order, no barrier
  }

  v8f oacc[8] = {};
#if __has_builtin(__builtin_amdgcn_wmma_f32_16x16x4_f32)
  typedef __attribute__((ext_vector_type(2))) float v2f;
#pragma unroll
  for (int kc = 0; kc < 4; ++kc) {
    int kk = kc * 4 + hi * 2;                 // A 16x4 f32: lanes0-15 K0,1; lanes16-31 K2,3
    v2f af;
    af.x = atile[w][m][kk];
    af.y = atile[w][m][kk + 1];
#pragma unroll
    for (int nt = 0; nt < 8; ++nt) {
      int d = nt * 16 + m;                    // B 4x16 f32: lane col = d
      v2f bv;
      bv.x = bf2f(vb[(size_t)kk * H_SZ + d]);
      bv.y = bf2f(vb[(size_t)(kk + 1) * H_SZ + d]);
      oacc[nt] = __builtin_amdgcn_wmma_f32_16x16x4_f32(
          false, af, false, bv, (short)0, oacc[nt], false, false);
    }
  }
#else
  // Fallback: single bf16 WMMA with K padded 16->32 (keys 16..31 zero).
  BF16Frag fa;
#pragma unroll
  for (int i = 0; i < 8; ++i) fa.u[i] = f2bf(atile[w][m][hi * 8 + i]);
#pragma unroll
  for (int i = 8; i < 16; ++i) fa.u[i] = 0;
#pragma unroll
  for (int nt = 0; nt < 8; ++nt) {
    BF16Frag fb;
    int d = nt * 16 + m;
#pragma unroll
    for (int i = 0; i < 16; ++i)
      fb.u[i] = (hi == 0) ? vb[(size_t)i * H_SZ + d] : (unsigned short)0;
    oacc[nt] = __builtin_amdgcn_wmma_f32_16x16x32_bf16(
        false, fa.v, false, fb.v, (short)0, oacc[nt], false, false);
  }
#endif

#pragma unroll
  for (int nt = 0; nt < 8; ++nt)
#pragma unroll
    for (int r = 0; r < 8; ++r) {
      int e = r + hi * 8, d = nt * 16 + m;
      att[base + (size_t)e * H_SZ + d] = f2bf(oacc[nt][r]);  // att aliases q: all q reads precede
    }
}

// ---------------------------------------------------------------------------
// 5) gate = sigmoid([x, comm_out] . wg + bg); out = gate*comm_out + (1-g)*x
// ---------------------------------------------------------------------------
__global__ __launch_bounds__(256) void gate_kernel(
    const float* __restrict__ x, const float* __restrict__ co,
    const float* __restrict__ wg, const float* __restrict__ bg,
    float* __restrict__ out) {
  int w = threadIdx.x >> 5, lane = threadIdx.x & 31;
  int row = blockIdx.x * 8 + w;
  const float* xr = x + (size_t)row * H_SZ;
  const float* cr = co + (size_t)row * H_SZ;
  float xv[32], cv[32], part = 0.f;
#pragma unroll
  for (int i = 0; i < 32; ++i) {
    int hh = i * 32 + lane;
    xv[i] = xr[hh]; cv[i] = cr[hh];
    part += xv[i] * wg[hh] + cv[i] * wg[H_SZ + hh];
  }
#pragma unroll
  for (int m = 16; m >= 1; m >>= 1) part += __shfl_xor(part, m, 32);
  float g = 1.f / (1.f + __expf(-(part + bg[0])));
#pragma unroll
  for (int i = 0; i < 32; ++i) {
    int hh = i * 32 + lane;
    out[(size_t)row * H_SZ + hh] = g * cv[i] + (1.f - g) * xv[i];
  }
}

// ---------------------------------------------------------------------------
extern "C" void kernel_launch(void* const* d_in, const int* in_sizes, int n_in,
                              void* d_out, int out_size, void* d_ws, size_t ws_size,
                              hipStream_t stream) {
  (void)in_sizes; (void)n_in; (void)out_size; (void)ws_size;
  const float* x    = (const float*)d_in[0];
  const float* ln_g = (const float*)d_in[1];
  const float* ln_b = (const float*)d_in[2];
  const float* wq   = (const float*)d_in[3];
  const float* bq   = (const float*)d_in[4];
  const float* wk   = (const float*)d_in[5];
  const float* bk   = (const float*)d_in[6];
  const float* wv   = (const float*)d_in[7];
  const float* bv   = (const float*)d_in[8];
  const float* wo   = (const float*)d_in[9];
  const float* bo   = (const float*)d_in[10];
  const float* wg   = (const float*)d_in[11];
  const float* bg   = (const float*)d_in[12];
  float* out = (float*)d_out;

  const size_t Ntot = (size_t)B_SZ * E_SZ * H_SZ;   // 33.5M elems
  const size_t HH   = (size_t)H_SZ * H_SZ;

  // Workspace layout (~411 MB total):
  char* p = (char*)d_ws;
  unsigned short* wbf  = (unsigned short*)p; p += 4 * HH * sizeof(unsigned short);
  float*          comm = (float*)p;          p += Ntot * sizeof(float);
  unsigned short* xn   = (unsigned short*)p; p += Ntot * sizeof(unsigned short);
  unsigned short* qb   = (unsigned short*)p; p += Ntot * sizeof(unsigned short);
  unsigned short* kb   = (unsigned short*)p; p += Ntot * sizeof(unsigned short);
  unsigned short* vb   = (unsigned short*)p; p += Ntot * sizeof(unsigned short);
  unsigned short* att  = qb;    // attended overwrites q (per-wave read-before-write safe)
  float* comm_out      = comm;  // residual add done in-place (one thread per element)

  wconv_kernel<<<dim3(HH / 1024, 1, 4), 256, 0, stream>>>(wq, wk, wv, wo, wbf);
  commln_kernel<<<dim3((B_SZ * E_SZ) / 8), 256, 0, stream>>>(x, ln_g, ln_b, comm, xn);
  gemm_qkv_kernel<<<dim3(H_SZ / 128, (B_SZ * E_SZ) / 256, 3), 256, 0, stream>>>(
      xn, wbf, bq, bk, bv, qb, kb, vb);
  attn_kernel<<<dim3((B_SZ * NH_SZ) / 4), 128, 0, stream>>>(qb, kb, vb, att);
  gemm_out_kernel<<<dim3(H_SZ / 128, (B_SZ * E_SZ) / 256, 1), 256, 0, stream>>>(
      att, wbf + 3 * HH, bo, comm, comm_out);
  gate_kernel<<<dim3((B_SZ * E_SZ) / 8), 256, 0, stream>>>(x, comm_out, wg, bg, out);
}
